// DecoderBlock_29248727286515
// MI455X (gfx1250) — matrix operand order repaired and checked by reading, not compiled, not run on hardware
//
#include <hip/hip_runtime.h>
#include <hip/hip_bf16.h>
#include <stdint.h>

// ---------------------------------------------------------------------------
// Types for CDNA5 WMMA (gfx1250, wave32)
// ---------------------------------------------------------------------------
typedef __attribute__((ext_vector_type(16))) __bf16 v16bf;
typedef __attribute__((ext_vector_type(8)))  float  v8f;

static __device__ inline v8f v8f_zero() {
    v8f z;
#pragma unroll
    for (int i = 0; i < 8; ++i) z[i] = 0.0f;
    return z;
}

// float -> bf16 round-to-nearest-even (storage-only use of __bf16)
static __device__ inline __bf16 f2bf(float f) {
    unsigned u = __float_as_uint(f);
    u = (u + 0x7FFFu + ((u >> 16) & 1u)) >> 16;
    unsigned short h = (unsigned short)u;
    __bf16 r;
    __builtin_memcpy(&r, &h, sizeof(r));
    return r;
}

// half-wave (16 lane) reductions: xor masks < 16 stay inside each half
static __device__ inline float hmax16(float v) {
#pragma unroll
    for (int m = 8; m >= 1; m >>= 1) v = fmaxf(v, __shfl_xor(v, m, 32));
    return v;
}
static __device__ inline float hsum16(float v) {
#pragma unroll
    for (int m = 8; m >= 1; m >>= 1) v += __shfl_xor(v, m, 32);
    return v;
}

// Fragment loaders: 128-bit accesses only.
// frag_split: elements 0..7 <- p[0..7], 8..15 <- p[16..23] (A layout after +8g bias)
static __device__ inline v16bf frag_split(const __bf16* p) {
    uint4 lo = *(const uint4*)p;
    uint4 hi = *(const uint4*)(p + 16);
    v16bf r;
    __builtin_memcpy(&r, &lo, 16);
    __builtin_memcpy((char*)&r + 16, &hi, 16);
    return r;
}
// frag_contig: 16 contiguous bf16 (k = 16g + s order after +16g bias; not used here)
static __device__ inline v16bf frag_contig(const __bf16* p) {
    uint4 lo = *(const uint4*)p;
    uint4 hi = *(const uint4*)(p + 8);
    v16bf r;
    __builtin_memcpy(&r, &lo, 16);
    __builtin_memcpy((char*)&r + 16, &hi, 16);
    return r;
}

// ---------------------------------------------------------------------------
// Async global -> LDS (CDNA5 GLOBAL_LOAD_ASYNC_TO_LDS_B128, ASYNCcnt-tracked),
// with synchronous fallback. Verified signature: (global int4*, local int4*,
// imm offset, imm cpol).
// ---------------------------------------------------------------------------
#if defined(__has_builtin)
#if __has_builtin(__builtin_amdgcn_global_load_async_to_lds_b128) && \
    __has_builtin(__builtin_amdgcn_s_wait_asynccnt)
#define HAVE_ASYNC_LDS 1
#endif
#endif

#if defined(HAVE_ASYNC_LDS)
typedef int v4i_gcc __attribute__((vector_size(16)));
typedef __attribute__((address_space(1))) v4i_gcc GV4;
typedef __attribute__((address_space(3))) v4i_gcc LV4;
static __device__ inline void stage16(void* lds_dst, const void* gsrc) {
    __builtin_amdgcn_global_load_async_to_lds_b128(
        (GV4*)(uintptr_t)gsrc,
        (LV4*)(unsigned)(uintptr_t)lds_dst, 0, 0);
}
static __device__ inline void stage_join() {
    __builtin_amdgcn_s_wait_asynccnt(0);
}
#else
static __device__ inline void stage16(void* lds_dst, const void* gsrc) {
    *(uint4*)lds_dst = *(const uint4*)gsrc;
}
static __device__ inline void stage_join() {}
#endif

// ---------------------------------------------------------------------------
// fp32 -> bf16 bulk convert
// ---------------------------------------------------------------------------
__global__ __launch_bounds__(256) void cvt_f32_bf16_kernel(
    const float* __restrict__ in, __bf16* __restrict__ out, long n) {
    long i = (long)blockIdx.x * blockDim.x + threadIdx.x;
    if (i < n) out[i] = f2bf(in[i]);
}

// ---------------------------------------------------------------------------
// Transpose + convert: fp32 [K x N] row-major -> bf16 [N x K] row-major
// ---------------------------------------------------------------------------
__global__ __launch_bounds__(256) void transpose_f32_bf16_kernel(
    const float* __restrict__ in, __bf16* __restrict__ out, int K, int N) {
    __shared__ float t[32][33];
    const int n0 = blockIdx.x * 32;
    const int k0 = blockIdx.y * 32;
    const int tx = threadIdx.x & 31;
    const int ty = threadIdx.x >> 5;
#pragma unroll
    for (int i = ty; i < 32; i += 8) t[i][tx] = in[(long)(k0 + i) * N + n0 + tx];
    __syncthreads();
#pragma unroll
    for (int i = ty; i < 32; i += 8)
        out[(long)(n0 + i) * K + k0 + tx] = f2bf(t[tx][i]);
}

// ---------------------------------------------------------------------------
// Tiled WMMA GEMM, double-buffered LDS + async staging software pipeline.
// C[M,N] = A[M,K] * B[K,N] (+bias)(+residual)(+gelu)
// A bf16 [M,K] row-major; BT bf16 [N,K] row-major (pre-transposed weights).
// Block: 256 threads (8 waves). Tile 128x128, BK=32. Wave tile 64x32.
// ---------------------------------------------------------------------------
#define BM 128
#define BN 128
#define BKD 32

template <bool RESID, bool GELU, bool OUT32, bool OUTB>
__global__ __launch_bounds__(256) void gemm_bf16_kernel(
    const __bf16* __restrict__ A, const __bf16* __restrict__ BT,
    const float* __restrict__ bias, const float* __restrict__ resid,
    float* __restrict__ out32, __bf16* __restrict__ outb,
    int M, int N, int K) {
    __shared__ __bf16 As[2][BM * BKD];    // [m][k], double-buffered
    __shared__ __bf16 BsT[2][BN * BKD];   // [n][k], double-buffered

    const int tid  = threadIdx.x;
    const int lane = tid & 31;
    const int wave = tid >> 5;
    const int g    = lane >> 4;
    const int ln   = lane & 15;
    const int wm   = (wave >> 2) * 64;
    const int wn   = (wave & 3) * 32;
    const long m0  = (long)blockIdx.y * BM;
    const long n0  = (long)blockIdx.x * BN;

    // staging: 512 16B chunks per tile, 2 unconditional chunks per thread
    const int r0 = tid >> 2, c0 = (tid & 3) * 8;              // chunk tid
    const int r1 = (tid + 256) >> 2, c1 = (tid & 3) * 8;      // chunk tid+256
    auto stage_tiles = [&](int k0, int buf) {
        stage16(As[buf] + r0 * BKD + c0, A + (m0 + r0) * (long)K + k0 + c0);
        stage16(As[buf] + r1 * BKD + c1, A + (m0 + r1) * (long)K + k0 + c1);
        stage16(BsT[buf] + r0 * BKD + c0, BT + (n0 + r0) * (long)K + k0 + c0);
        stage16(BsT[buf] + r1 * BKD + c1, BT + (n0 + r1) * (long)K + k0 + c1);
    };

    v8f acc[4][2];
#pragma unroll
    for (int i = 0; i < 4; ++i)
#pragma unroll
        for (int j = 0; j < 2; ++j) acc[i][j] = v8f_zero();

    const int nk = K / BKD;
    stage_tiles(0, 0);
    stage_join();
    __syncthreads();

    for (int kk = 0; kk < nk; ++kk) {
        const int cur = kk & 1;
        // prefetch next tile into the other buffer (async, overlaps WMMAs)
        if (kk + 1 < nk) stage_tiles((kk + 1) * BKD, cur ^ 1);

        v16bf af[4], bfg[2];
#pragma unroll
        for (int i = 0; i < 4; ++i)
            af[i] = frag_split(As[cur] + (wm + i * 16 + ln) * BKD + 8 * g);
#pragma unroll
        for (int j = 0; j < 2; ++j)
            bfg[j] = frag_split(BsT[cur] + (wn + j * 16 + ln) * BKD + 8 * g);
#pragma unroll
        for (int i = 0; i < 4; ++i)
#pragma unroll
            for (int j = 0; j < 2; ++j)
                acc[i][j] = __builtin_amdgcn_wmma_f32_16x16x32_bf16(
                    false, af[i], false, bfg[j], (short)0, acc[i][j], false, false);

        stage_join();       // async copies have been running under the WMMAs
        __syncthreads();    // all waves done computing cur & staging cur^1
    }

    // epilogue  (C layout: M = r + 8g, N = lane&15)
#pragma unroll
    for (int i = 0; i < 4; ++i) {
#pragma unroll
        for (int j = 0; j < 2; ++j) {
            long col = n0 + wn + j * 16 + ln;
            float bv = bias[col];
#pragma unroll
            for (int r = 0; r < 8; ++r) {
                long row = m0 + wm + i * 16 + r + 8 * g;
                long idx = row * (long)N + col;
                float v = acc[i][j][r] + bv;
                if (RESID) v += resid[idx];
                if (GELU) v = 0.5f * v * (1.0f + erff(v * 0.70710678118654752f));
                if (OUT32) out32[idx] = v;
                if (OUTB)  outb[idx]  = f2bf(v);
            }
        }
    }
}

// ---------------------------------------------------------------------------
// Flash attention (causal). Block = 64 query rows x 1 head x 1 batch.
// 128 threads = 4 waves; each wave owns 16 query rows.
// ---------------------------------------------------------------------------
__global__ __launch_bounds__(128) void attn_kernel(
    const __bf16* __restrict__ Qb, const __bf16* __restrict__ Kb,
    const __bf16* __restrict__ Vb, __bf16* __restrict__ Ob) {
    const int S = 2048, HD = 768, DKC = 64;
    __shared__ __bf16 Ks[32 * 64];    // [kv][d]
    __shared__ __bf16 VsT[64 * 32];   // [d][kv] (transposed for contiguous B-frags)
    __shared__ __bf16 Ps[4][16 * 32];

    const int tid  = threadIdx.x;
    const int lane = tid & 31;
    const int wave = tid >> 5;
    const int g    = lane >> 4;
    const int ln   = lane & 15;
    const int qblk = blockIdx.x * 64;
    const int h    = blockIdx.y;
    const int b    = blockIdx.z;
    const long base = ((long)b * S) * HD + h * DKC;

    // preload Q fragments (128-bit loads)
    const __bf16* qp = Qb + base + (long)(qblk + wave * 16 + ln) * HD;
    v16bf aq0 = frag_split(qp + 8 * g);
    v16bf aq1 = frag_split(qp + 32 + 8 * g);

    v8f accO[4];
#pragma unroll
    for (int i = 0; i < 4; ++i) accO[i] = v8f_zero();
    float mrow[8], lrow[8];
#pragma unroll
    for (int r = 0; r < 8; ++r) { mrow[r] = -1e30f; lrow[r] = 0.0f; }

    // K staging chunks for this thread (2 unconditional per tile)
    const int kr0 = tid >> 3, kc0 = (tid & 7) * 8;
    const int kr1 = (tid + 128) >> 3, kc1 = (tid & 7) * 8;

    const int kvend = qblk + 64;  // causal: skip fully-masked tiles
    for (int kv0 = 0; kv0 < kvend; kv0 += 32) {
        __syncthreads();
        // K tile: async coalesced copy, row-major [kv][d]
        stage16(Ks + kr0 * 64 + kc0, Kb + base + (long)(kv0 + kr0) * HD + kc0);
        stage16(Ks + kr1 * 64 + kc1, Kb + base + (long)(kv0 + kr1) * HD + kc1);
        // V tile: transpose into VsT[d][kv], packing bf16 pairs into u32 stores
        {
            int pi = tid >> 3;       // kv pair 0..15
            int dc = tid & 7;        // d chunk 0..7
            int kv = 2 * pi;
            int d0 = dc * 8;
            const __bf16* v0p = Vb + base + (long)(kv0 + kv) * HD + d0;
            uint4 a = *(const uint4*)v0p;
            uint4 b2 = *(const uint4*)(v0p + HD);
            const unsigned short* pa = (const unsigned short*)&a;
            const unsigned short* pb = (const unsigned short*)&b2;
#pragma unroll
            for (int e = 0; e < 8; ++e) {
                unsigned pk = (unsigned)pa[e] | ((unsigned)pb[e] << 16);
                *(unsigned*)(VsT + (d0 + e) * 32 + kv) = pk;
            }
        }
        stage_join();
        __syncthreads();

        // scores: S = Q(16x64) * K^T(64x32) -> two 16x16 accumulators
        v8f sc[2];
#pragma unroll
        for (int ns = 0; ns < 2; ++ns) {
            const __bf16* kp = Ks + (16 * ns + ln) * 64;
            v16bf bk0 = frag_split(kp + 8 * g);
            v16bf bk1 = frag_split(kp + 32 + 8 * g);
            v8f s = v8f_zero();
            s = __builtin_amdgcn_wmma_f32_16x16x32_bf16(false, aq0, false, bk0,
                                                        (short)0, s, false, false);
            s = __builtin_amdgcn_wmma_f32_16x16x32_bf16(false, aq1, false, bk1,
                                                        (short)0, s, false, false);
            sc[ns] = s;
        }
        // scale + causal mask
#pragma unroll
        for (int ns = 0; ns < 2; ++ns) {
            int keyg = kv0 + 16 * ns + ln;
#pragma unroll
            for (int r = 0; r < 8; ++r) {
                int qg = qblk + wave * 16 + r + 8 * g;
                float v = sc[ns][r] * 0.125f;
                sc[ns][r] = (keyg > qg) ? -1e9f : v;
            }
        }
        // online softmax
        float p0[8], p1[8];
#pragma unroll
        for (int r = 0; r < 8; ++r) {
            float t = hmax16(fmaxf(sc[0][r], sc[1][r]));
            float mnew = fmaxf(mrow[r], t);
            p0[r] = __expf(sc[0][r] - mnew);
            p1[r] = __expf(sc[1][r] - mnew);
            float srow = hsum16(p0[r] + p1[r]);
            float corr = __expf(mrow[r] - mnew);
            lrow[r] = lrow[r] * corr + srow;
            mrow[r] = mnew;
#pragma unroll
            for (int nd = 0; nd < 4; ++nd) accO[nd][r] *= corr;
        }
        // reshape P (C layout) -> A layout via per-wave LDS
#pragma unroll
        for (int r = 0; r < 8; ++r) {
            Ps[wave][(r + 8 * g) * 32 + ln]      = f2bf(p0[r]);
            Ps[wave][(r + 8 * g) * 32 + 16 + ln] = f2bf(p1[r]);
        }
        v16bf ap = frag_split(&Ps[wave][ln * 32 + 8 * g]);
        // O += P(16x32) * V(32x64); V B-frags split (table order) from VsT
#pragma unroll
        for (int nd = 0; nd < 4; ++nd) {
            v16bf bv = frag_split(VsT + (nd * 16 + ln) * 32 + 8 * g);
            accO[nd] = __builtin_amdgcn_wmma_f32_16x16x32_bf16(
                false, ap, false, bv, (short)0, accO[nd], false, false);
        }
    }

    // normalize + store ctx (bf16, [B,S,H*DK])
#pragma unroll
    for (int r = 0; r < 8; ++r) {
        float inv = 1.0f / lrow[r];
        int row = qblk + wave * 16 + r + 8 * g;
#pragma unroll
        for (int nd = 0; nd < 4; ++nd) {
            Ob[base + (long)row * HD + nd * 16 + ln] = f2bf(accO[nd][r] * inv);
        }
    }
}

// ---------------------------------------------------------------------------
// LayerNorm over rows of 768; one block (256 threads) per row.
// ---------------------------------------------------------------------------
__global__ __launch_bounds__(256) void ln_kernel(
    const float* __restrict__ in, const float* __restrict__ gamma,
    const float* __restrict__ beta, float* __restrict__ out32,
    __bf16* __restrict__ outb) {
    const int D = 768;
    __shared__ float sbuf[256];
    const int tid = threadIdx.x;
    const long row = blockIdx.x;
    const float* p = in + row * D;

    float x0 = p[tid], x1 = p[tid + 256], x2 = p[tid + 512];
    sbuf[tid] = x0 + x1 + x2;
    __syncthreads();
#pragma unroll
    for (int o = 128; o > 0; o >>= 1) {
        if (tid < o) sbuf[tid] += sbuf[tid + o];
        __syncthreads();
    }
    float mu = sbuf[0] * (1.0f / D);
    __syncthreads();

    float d0 = x0 - mu, d1 = x1 - mu, d2 = x2 - mu;
    sbuf[tid] = d0 * d0 + d1 * d1 + d2 * d2;
    __syncthreads();
#pragma unroll
    for (int o = 128; o > 0; o >>= 1) {
        if (tid < o) sbuf[tid] += sbuf[tid + o];
        __syncthreads();
    }
    float inv = rsqrtf(sbuf[0] * (1.0f / D) + 1e-5f);

    float y0 = d0 * inv * gamma[tid]       + beta[tid];
    float y1 = d1 * inv * gamma[tid + 256] + beta[tid + 256];
    float y2 = d2 * inv * gamma[tid + 512] + beta[tid + 512];
    float* q = out32 + row * D;
    q[tid] = y0; q[tid + 256] = y1; q[tid + 512] = y2;
    if (outb) {
        __bf16* qb = outb + row * D;
        qb[tid] = f2bf(y0); qb[tid + 256] = f2bf(y1); qb[tid + 512] = f2bf(y2);
    }
}

// ---------------------------------------------------------------------------
// Host launcher
// ---------------------------------------------------------------------------
extern "C" void kernel_launch(void* const* d_in, const int* in_sizes, int n_in,
                              void* d_out, int out_size, void* d_ws, size_t ws_size,
                              hipStream_t stream) {
    const int Bb = 4, S = 2048, D = 768, DF = 3072;
    const long MS = (long)Bb * S;  // 8192

    const float* x   = (const float*)d_in[0];
    const float* wq  = (const float*)d_in[2];
    const float* bq  = (const float*)d_in[3];
    const float* wk  = (const float*)d_in[4];
    const float* bk  = (const float*)d_in[5];
    const float* wv  = (const float*)d_in[6];
    const float* bv  = (const float*)d_in[7];
    const float* wo  = (const float*)d_in[8];
    const float* bo  = (const float*)d_in[9];
    const float* g1  = (const float*)d_in[10];
    const float* be1 = (const float*)d_in[11];
    const float* w1  = (const float*)d_in[12];
    const float* b1  = (const float*)d_in[13];
    const float* w2  = (const float*)d_in[14];
    const float* b2  = (const float*)d_in[15];
    const float* g2  = (const float*)d_in[16];
    const float* be2 = (const float*)d_in[17];
    float* out = (float*)d_out;

    char* wsp = (char*)d_ws;
    auto carve = [&](size_t bytes) -> void* {
        void* p = (void*)wsp;
        wsp += (bytes + 255) & ~(size_t)255;
        return p;
    };
    __bf16* xb   = (__bf16*)carve(MS * D * 2);
    __bf16* wqT  = (__bf16*)carve((size_t)D * D * 2);    // [N=768][K=768]
    __bf16* wkT  = (__bf16*)carve((size_t)D * D * 2);
    __bf16* wvT  = (__bf16*)carve((size_t)D * D * 2);
    __bf16* woT  = (__bf16*)carve((size_t)D * D * 2);
    __bf16* w1T  = (__bf16*)carve((size_t)D * DF * 2);   // [3072][768]
    __bf16* w2T  = (__bf16*)carve((size_t)DF * D * 2);   // [768][3072]
    __bf16* qb   = (__bf16*)carve(MS * D * 2);
    __bf16* kb_  = (__bf16*)carve(MS * D * 2);
    __bf16* vb_  = (__bf16*)carve(MS * D * 2);
    __bf16* ctxb = (__bf16*)carve(MS * D * 2);
    __bf16* hb   = (__bf16*)carve(MS * D * 2);
    __bf16* fb   = (__bf16*)carve(MS * DF * 2);
    float*  t1   = (float*)carve(MS * D * 4);
    float*  hh   = (float*)carve(MS * D * 4);

    // activations: straight convert; weights: transpose+convert ([K,N] -> [N,K])
    cvt_f32_bf16_kernel<<<(unsigned)((MS * D + 255) / 256), 256, 0, stream>>>(x, xb, MS * D);
    auto tr = [&](const float* src, __bf16* dst, int K, int N) {
        transpose_f32_bf16_kernel<<<dim3(N / 32, K / 32), 256, 0, stream>>>(src, dst, K, N);
    };
    tr(wq, wqT, D, D);
    tr(wk, wkT, D, D);
    tr(wv, wvT, D, D);
    tr(wo, woT, D, D);
    tr(w1, w1T, D, DF);
    tr(w2, w2T, DF, D);

    dim3 blk(256);
    dim3 grid_d(D / BN, (unsigned)(MS / BM));    // 6 x 64
    dim3 grid_f(DF / BN, (unsigned)(MS / BM));   // 24 x 64

    // QKV projections -> bf16
    gemm_bf16_kernel<false, false, false, true><<<grid_d, blk, 0, stream>>>(
        xb, wqT, bq, nullptr, nullptr, qb, (int)MS, D, D);
    gemm_bf16_kernel<false, false, false, true><<<grid_d, blk, 0, stream>>>(
        xb, wkT, bk, nullptr, nullptr, kb_, (int)MS, D, D);
    gemm_bf16_kernel<false, false, false, true><<<grid_d, blk, 0, stream>>>(
        xb, wvT, bv, nullptr, nullptr, vb_, (int)MS, D, D);

    // causal flash attention
    attn_kernel<<<dim3(S / 64, 12, Bb), dim3(128), 0, stream>>>(qb, kb_, vb_, ctxb);

    // output projection + residual(x) -> t1 ; LN1 -> h (fp32 + bf16)
    gemm_bf16_kernel<true, false, true, false><<<grid_d, blk, 0, stream>>>(
        ctxb, woT, bo, x, t1, nullptr, (int)MS, D, D);
    ln_kernel<<<(unsigned)MS, 256, 0, stream>>>(t1, g1, be1, hh, hb);

    // FFN: gelu(h@w1+b1) -> fb ; fb@w2+b2 + h -> t1 ; LN2 -> out
    gemm_bf16_kernel<false, true, false, true><<<grid_f, blk, 0, stream>>>(
        hb, w1T, b1, nullptr, nullptr, fb, (int)MS, DF, D);
    gemm_bf16_kernel<true, false, true, false><<<grid_d, blk, 0, stream>>>(
        fb, w2T, b2, hh, t1, nullptr, (int)MS, D, DF);
    ln_kernel<<<(unsigned)MS, 256, 0, stream>>>(t1, g2, be2, out, nullptr);
}
